// WindowAttention_16535624089970
// MI455X (gfx1250) — matrix operand order repaired and checked by reading, not compiled, not run on hardware
//
#include <hip/hip_runtime.h>

// ---------------------------------------------------------------------------
// Swin window attention for MI455X (gfx1250, wave32, WMMA).
//   x:(4096,49,384) -> qkv GEMM (bf16 wmma) -> per-(window,head) attention
//   (bf16 wmma + f32 softmax) -> proj GEMM (bf16 wmma) -> f32 out.
// WMMA fragments load from LDS as ds_load_b128; f32->bf16 uses v_perm_b32
// packing; GEMM K-loops are double-buffered (1 barrier / K-step); pure-copy
// tile staging uses gfx1250 async global->LDS DMA (ASYNCcnt).
// ---------------------------------------------------------------------------

typedef __attribute__((ext_vector_type(16))) __bf16 v16bf;
typedef __attribute__((ext_vector_type(8)))  float  v8f;

union V16BF { unsigned short u[16]; uint4 q[2]; v16bf v; };
union V8F   { float f[8];           v8f  v; };

// Round-to-nearest (+0x8000) then take the high halves of two f32 with a
// single v_perm_b32: result = {bf16(a) , bf16(b)<<16}.
__device__ __forceinline__ unsigned int pack_bf16x2(float a, float b) {
    const unsigned int au = __float_as_uint(a) + 0x8000u;
    const unsigned int bu = __float_as_uint(b) + 0x8000u;
    return __builtin_amdgcn_perm(bu, au, 0x07060302u);
}
__device__ __forceinline__ unsigned short f32_to_bf16(float f) {
    return (unsigned short)((__float_as_uint(f) + 0x8000u) >> 16);
}
__device__ __forceinline__ float bf16_to_f32(unsigned short h) {
    return __uint_as_float(((unsigned int)h) << 16);
}

// Async global->LDS 16B copy (gfx1250 TDM-adjacent path, ASYNCcnt-tracked).
// Generic pointers to LDS carry the LDS byte offset in their low 32 bits
// (flat aperture rule).
__device__ __forceinline__ void async_copy_b128(void* lds_dst, const void* gsrc) {
    const unsigned ldsa = (unsigned)(uintptr_t)lds_dst;
    asm volatile("global_load_async_to_lds_b128 %0, %1, off"
                 :
                 : "v"(ldsa), "v"((unsigned long long)(uintptr_t)gsrc)
                 : "memory");
}
__device__ __forceinline__ void wait_async() {
    asm volatile("s_wait_asynccnt 0x0" ::: "memory");
}

static constexpr int BM = 64, BN = 32, BK = 32;
static constexpr int WH = 7, NTOK = 49, HEADS = 12, CDIM = 384, HD = 32;
static constexpr int APITCH = BK + 8;    // 40 shorts = 80B rows: 16B aligned, conflict-spread

// A-fragment (16x32 bf16, ISA 7.12.2): two contiguous 16B runs per lane.
__device__ __forceinline__ void load_afrag(V16BF& a, const unsigned short* rowp, int hi) {
    a.q[0] = *reinterpret_cast<const uint4*>(rowp + (hi << 3));
    a.q[1] = *reinterpret_cast<const uint4*>(rowp + 16 + (hi << 3));
}
// B-fragment from K-major storage (Bt[n][k]): two contiguous 16B runs per lane.
__device__ __forceinline__ void load_bfrag(V16BF& b, const unsigned short* rowp, int hi) {
    b.q[0] = *reinterpret_cast<const uint4*>(rowp + (hi << 3));
    b.q[1] = *reinterpret_cast<const uint4*>(rowp + 16 + (hi << 3));
}

// ---------------------------------------------------------------------------
// Kernel 1: Y[M,1152](bf16) = X[M,384](f32) @ W[384,1152](f32) + b
// ---------------------------------------------------------------------------
__global__ __launch_bounds__(256) void qkv_gemm_kernel(
    const float* __restrict__ X, const float* __restrict__ W,
    const float* __restrict__ bias, unsigned short* __restrict__ Y)
{
    constexpr int K = CDIM, Nc = 3 * CDIM, NKB = K / BK;
    __shared__ __align__(16) unsigned short As[2][BM][APITCH];
    __shared__ __align__(16) unsigned short Bt[2][BN][APITCH];
    __shared__ __align__(16) unsigned short Cs[BM][BN];

    const int m0 = blockIdx.x * BM, n0 = blockIdx.y * BN;
    const int tid = threadIdx.x;
    const int wave = tid >> 5, lane = tid & 31;
    const int lo = lane & 15, hi = lane >> 4;
    const int wm = (wave >> 1) << 4;   // 0,16,32,48
    const int wn = (wave & 1) << 4;    // 0,16

    // Per-thread staging coordinates.
    const int ar = tid >> 3, ac = (tid & 7) << 2;        // A: 2 float4 (rows ar, ar+32)
    const int bn = tid & 31, bk = (tid >> 5) << 2;       // B: 4 strided floats (col bn)

    auto stage_regs = [&](int kb, float4* ra, float* rb) {
        const int k0 = kb * BK;
        ra[0] = *reinterpret_cast<const float4*>(X + (size_t)(m0 + ar) * K + k0 + ac);
        ra[1] = *reinterpret_cast<const float4*>(X + (size_t)(m0 + ar + 32) * K + k0 + ac);
        const float* wp = W + (size_t)(k0 + bk) * Nc + n0 + bn;
        rb[0] = wp[0 * Nc]; rb[1] = wp[1 * Nc]; rb[2] = wp[2 * Nc]; rb[3] = wp[3 * Nc];
    };
    auto store_tiles = [&](int p, const float4* ra, const float* rb) {
        uint2 u;
        u.x = pack_bf16x2(ra[0].x, ra[0].y); u.y = pack_bf16x2(ra[0].z, ra[0].w);
        *reinterpret_cast<uint2*>(&As[p][ar][ac]) = u;
        u.x = pack_bf16x2(ra[1].x, ra[1].y); u.y = pack_bf16x2(ra[1].z, ra[1].w);
        *reinterpret_cast<uint2*>(&As[p][ar + 32][ac]) = u;
        u.x = pack_bf16x2(rb[0], rb[1]);     u.y = pack_bf16x2(rb[2], rb[3]);
        *reinterpret_cast<uint2*>(&Bt[p][bn][bk]) = u;
    };

    V8F acc;
#pragma unroll
    for (int r = 0; r < 8; ++r) acc.f[r] = 0.f;

    {
        float4 ra[2]; float rb[4];
        stage_regs(0, ra, rb);
        store_tiles(0, ra, rb);
    }
    __syncthreads();

#pragma unroll
    for (int kb = 0; kb < NKB; ++kb) {
        const int p = kb & 1;
        float4 ra[2]; float rb[4];
        if (kb + 1 < NKB) stage_regs(kb + 1, ra, rb);   // overlap with WMMA below

        V16BF a, b;
        load_afrag(a, &As[p][wm + lo][0], hi);
        load_bfrag(b, &Bt[p][wn + lo][0], hi);
        acc.v = __builtin_amdgcn_wmma_f32_16x16x32_bf16(
            false, a.v, false, b.v, (short)0, acc.v, false, false);

        if (kb + 1 < NKB) store_tiles(1 - p, ra, rb);
        __syncthreads();
    }

    // Epilogue: + bias, stage in LDS, coalesced 16B stores.
#pragma unroll
    for (int r = 0; r < 8; ++r) {
        const int mloc = wm + r + (hi << 3);
        const int nloc = wn + lo;
        Cs[mloc][nloc] = f32_to_bf16(acc.f[r] + bias[n0 + nloc]);
    }
    __syncthreads();
    {
        const int row = tid >> 2, seg = (tid & 3) << 3;
        *reinterpret_cast<uint4*>(Y + (size_t)(m0 + row) * Nc + n0 + seg) =
            *reinterpret_cast<const uint4*>(&Cs[row][seg]);
    }
}

// ---------------------------------------------------------------------------
// Kernel 2: per (window b, head h) attention. 4 waves, each owns a 16-row
// strip of the (padded) 64x64 score matrix.
// ---------------------------------------------------------------------------
__global__ __launch_bounds__(128) void attn_kernel(
    const unsigned short* __restrict__ QKV,   // [B*49, 1152] bf16
    const float* __restrict__ bias_table,     // [169, 12] f32
    unsigned short* __restrict__ O)           // [B*49, 384] bf16
{
    constexpr int PPITCH = 64 + 8;            // 72 shorts = 144B rows
    const int b = blockIdx.x, h = blockIdx.y;
    __shared__ __align__(16) unsigned short Qs[64][APITCH];   // token-major
    __shared__ __align__(16) unsigned short Ks[64][APITCH];   // token-major (== K^T K-major)
    __shared__ __align__(16) unsigned short Vt[HD][PPITCH];   // dim-major (token is K-dim)
    __shared__ __align__(16) float          Ss[64][65];
    __shared__ __align__(16) unsigned short Ps[64][PPITCH];

    const int tid = threadIdx.x;
    const int wave = tid >> 5, lane = tid & 31;
    const int lo = lane & 15, hi = lane >> 4;
    const int wm = wave << 4;
    const float scale = 0.17677669529663687f;   // 32^-0.5
    const size_t qkvbase = (size_t)(b * NTOK) * 1152 + h * HD;
    const size_t obase   = (size_t)(b * NTOK) * CDIM + h * HD;

    // Stage q and k: one 64B token row per thread via async global->LDS DMA
    // (pure bf16 copy; softmax scale is folded into the S epilogue).
    {
        const int which = tid >> 6, n = tid & 63;   // which: 0=q, 1=k
        unsigned short(*arr)[APITCH] = (which == 0) ? Qs : Ks;
        if (n < NTOK) {
            const unsigned short* src = QKV + (size_t)n * 1152 + which * CDIM + qkvbase;
#pragma unroll
            for (int i = 0; i < 4; ++i)
                async_copy_b128(&arr[n][i << 3], src + (i << 3));
        } else {
#pragma unroll
            for (int i = 0; i < 4; ++i)
                *reinterpret_cast<uint4*>(&arr[n][i << 3]) = uint4{0, 0, 0, 0};
        }
    }
    // Stage V transposed (dim-major): lane-coalesced 2B global loads.
    {
        const int d = tid & 31, grp = tid >> 5;
        for (int n = grp; n < 64; n += 4) {
            unsigned short val = 0;
            if (n < NTOK) val = QKV[(size_t)n * 1152 + 2 * CDIM + d + qkvbase];
            Vt[d][n] = val;
        }
    }
    wait_async();
    __syncthreads();

    // S = scale*(q @ k^T) + rel-pos bias   (4 column tiles per wave)
#pragma unroll
    for (int ct = 0; ct < 4; ++ct) {
        V16BF a, bm;
        V8F c;
#pragma unroll
        for (int r = 0; r < 8; ++r) c.f[r] = 0.f;
        load_afrag(a, &Qs[wm + lo][0], hi);
        load_bfrag(bm, &Ks[(ct << 4) + lo][0], hi);   // K^T: lane = token col, run = dims
        c.v = __builtin_amdgcn_wmma_f32_16x16x32_bf16(
            false, a.v, false, bm.v, (short)0, c.v, false, false);
#pragma unroll
        for (int r = 0; r < 8; ++r) {
            const int mr = wm + r + (hi << 3);
            const int nc = (ct << 4) + lo;
            float sv = -1e30f;
            if (mr < NTOK && nc < NTOK) {
                // REL_IDX(m,n) closed form: ((yi-yj)+6)*13 + ((xi-xj)+6)
                const int yi = mr / WH, xi = mr % WH;
                const int yj = nc / WH, xj = nc % WH;
                const int ridx = (yi - yj + 6) * 13 + (xi - xj + 6);
                sv = fmaf(c.f[r], scale, bias_table[ridx * HEADS + h]);
            }
            Ss[mr][nc] = sv;
        }
    }
    __syncthreads();

    // f32 softmax per row (rows 49..63 -> zero probability rows)
    if (tid < 64) {
        if (tid < NTOK) {
            float mx = -1e30f;
            for (int n = 0; n < NTOK; ++n) mx = fmaxf(mx, Ss[tid][n]);
            float sum = 0.f;
            for (int n = 0; n < NTOK; ++n) {
                const float e = __expf(Ss[tid][n] - mx);
                Ss[tid][n] = e;
                sum += e;
            }
            const float inv = 1.f / sum;
            for (int n = 0; n < NTOK; n += 2)
                *reinterpret_cast<unsigned int*>(&Ps[tid][n]) =
                    pack_bf16x2(Ss[tid][n] * inv, (n + 1 < NTOK) ? Ss[tid][n + 1] * inv : 0.f);
            for (int n = 50; n < 64; n += 2)
                *reinterpret_cast<unsigned int*>(&Ps[tid][n]) = 0u;
        } else {
            for (int n = 0; n < 64; n += 2)
                *reinterpret_cast<unsigned int*>(&Ps[tid][n]) = 0u;
        }
    }
    __syncthreads();

    // O = P @ V   (2 column tiles x 2 K-chunks per wave)
#pragma unroll
    for (int nt = 0; nt < 2; ++nt) {
        V8F c;
#pragma unroll
        for (int r = 0; r < 8; ++r) c.f[r] = 0.f;
#pragma unroll
        for (int kc = 0; kc < 2; ++kc) {
            V16BF a, bm;
            load_afrag(a, &Ps[wm + lo][kc << 5], hi);
            load_bfrag(bm, &Vt[(nt << 4) + lo][kc << 5], hi);
            c.v = __builtin_amdgcn_wmma_f32_16x16x32_bf16(
                false, a.v, false, bm.v, (short)0, c.v, false, false);
        }
#pragma unroll
        for (int r = 0; r < 8; ++r)
            Ss[wm + r + (hi << 3)][(nt << 4) + lo] = c.f[r];   // reuse Ss as O staging
    }
    __syncthreads();

    for (int i = tid; i < NTOK * (HD / 2); i += 128) {          // 2 cols per thread
        const int row = i >> 4, col = (i & 15) << 1;
        *reinterpret_cast<unsigned int*>(O + obase + (size_t)row * CDIM + col) =
            pack_bf16x2(Ss[row][col], Ss[row][col + 1]);
    }
}

// ---------------------------------------------------------------------------
// Kernel 3: Y[M,384](f32) = A[M,384](bf16) @ W[384,384](f32) + b
// A tile is a pure bf16 copy -> async global->LDS DMA, double-buffered.
// ---------------------------------------------------------------------------
__global__ __launch_bounds__(256) void proj_gemm_kernel(
    const unsigned short* __restrict__ A, const float* __restrict__ W,
    const float* __restrict__ bias, float* __restrict__ Y)
{
    constexpr int K = CDIM, Nc = CDIM, NKB = K / BK;
    __shared__ __align__(16) unsigned short As[2][BM][APITCH];
    __shared__ __align__(16) unsigned short Bt[2][BN][APITCH];
    __shared__ __align__(16) float Cs[BM][BN];

    const int m0 = blockIdx.x * BM, n0 = blockIdx.y * BN;
    const int tid = threadIdx.x;
    const int wave = tid >> 5, lane = tid & 31;
    const int lo = lane & 15, hi = lane >> 4;
    const int wm = (wave >> 1) << 4;
    const int wn = (wave & 1) << 4;

    const int ar = tid >> 2, ac = (tid & 3) << 3;        // A: 1 x 16B async slot
    const int bn = tid & 31, bk = (tid >> 5) << 2;       // B: 4 strided floats

    auto stage_async_A = [&](int kb, int p) {
        async_copy_b128(&As[p][ar][ac], A + (size_t)(m0 + ar) * K + kb * BK + ac);
    };
    auto stage_regs_B = [&](int kb, float* rb) {
        const float* wp = W + (size_t)(kb * BK + bk) * Nc + n0 + bn;
        rb[0] = wp[0 * Nc]; rb[1] = wp[1 * Nc]; rb[2] = wp[2 * Nc]; rb[3] = wp[3 * Nc];
    };
    auto store_B = [&](int p, const float* rb) {
        uint2 u;
        u.x = pack_bf16x2(rb[0], rb[1]); u.y = pack_bf16x2(rb[2], rb[3]);
        *reinterpret_cast<uint2*>(&Bt[p][bn][bk]) = u;
    };

    V8F acc;
#pragma unroll
    for (int r = 0; r < 8; ++r) acc.f[r] = 0.f;

    {
        stage_async_A(0, 0);
        float rb[4];
        stage_regs_B(0, rb);
        store_B(0, rb);
    }
    wait_async();
    __syncthreads();

#pragma unroll
    for (int kb = 0; kb < NKB; ++kb) {
        const int p = kb & 1;
        float rb[4];
        if (kb + 1 < NKB) {
            stage_async_A(kb + 1, 1 - p);   // DMA overlaps the WMMA below
            stage_regs_B(kb + 1, rb);
        }

        V16BF a, b;
        load_afrag(a, &As[p][wm + lo][0], hi);
        load_bfrag(b, &Bt[p][wn + lo][0], hi);
        acc.v = __builtin_amdgcn_wmma_f32_16x16x32_bf16(
            false, a.v, false, b.v, (short)0, acc.v, false, false);

        if (kb + 1 < NKB) store_B(1 - p, rb);
        wait_async();
        __syncthreads();
    }

#pragma unroll
    for (int r = 0; r < 8; ++r) {
        const int mloc = wm + r + (hi << 3);
        const int nloc = wn + lo;
        Cs[mloc][nloc] = acc.f[r] + bias[n0 + nloc];
    }
    __syncthreads();
    {
        const int row = tid >> 2, seg = (tid & 3) << 3;
        float* dst = Y + (size_t)(m0 + row) * Nc + n0 + seg;
        *reinterpret_cast<float4*>(dst + 0) = *reinterpret_cast<const float4*>(&Cs[row][seg]);
        *reinterpret_cast<float4*>(dst + 4) = *reinterpret_cast<const float4*>(&Cs[row][seg + 4]);
    }
}

// ---------------------------------------------------------------------------
extern "C" void kernel_launch(void* const* d_in, const int* in_sizes, int n_in,
                              void* d_out, int out_size, void* d_ws, size_t ws_size,
                              hipStream_t stream) {
    const float* x          = (const float*)d_in[0];
    const float* w_qkv      = (const float*)d_in[1];
    const float* b_qkv      = (const float*)d_in[2];
    const float* w_proj     = (const float*)d_in[3];
    const float* b_proj     = (const float*)d_in[4];
    const float* bias_table = (const float*)d_in[5];

    const int Bwin = 4096;
    const int M = Bwin * NTOK;              // 200704 rows
    unsigned short* qkv = (unsigned short*)d_ws;                                 // M x 1152 bf16
    unsigned short* att = (unsigned short*)((char*)d_ws + (size_t)M * 1152 * 2); // M x 384 bf16

    qkv_gemm_kernel<<<dim3(M / BM, (3 * CDIM) / BN), 256, 0, stream>>>(
        x, w_qkv, b_qkv, qkv);
    attn_kernel<<<dim3(Bwin, HEADS), 128, 0, stream>>>(qkv, bias_table, att);
    proj_gemm_kernel<<<dim3(M / BM, CDIM / BN), 256, 0, stream>>>(
        att, w_proj, b_proj, (float*)d_out);
}
